// ST_LSTM_85117661872157
// MI455X (gfx1250) — matrix operand order repaired and checked by reading, not compiled, not run on hardware
//
#include <hip/hip_runtime.h>

typedef __attribute__((ext_vector_type(16))) __bf16 v16bf;
typedef __attribute__((ext_vector_type(8)))  float  v8f;
typedef __attribute__((ext_vector_type(4)))  float  v4f;

namespace {
constexpr int T_ = 64, J_ = 25, B_ = 64, D_ = 128, H_ = 512;
constexpr int NG_ = 5 * H_;          // 2560 gate columns
constexpr int NT_ = NG_ / 16;        // 160 gate column tiles
constexpr int KT_D = D_ / 32;        // 4  K-tiles for x
constexpr int KT_H = H_ / 32;        // 16 K-tiles for h

constexpr size_t TILE_B = 1024;      // bytes per 16x32 bf16 tile (= 16x16 f32 tile)
constexpr size_t O_WIH0  = 0;
constexpr size_t SZ_WIH0 = (size_t)NT_ * KT_D * TILE_B;
constexpr size_t SZ_WH   = (size_t)NT_ * KT_H * TILE_B;
constexpr size_t O_WHHT0 = O_WIH0  + SZ_WIH0;
constexpr size_t O_WHHS0 = O_WHHT0 + SZ_WH;
constexpr size_t O_WIH1  = O_WHHS0 + SZ_WH;
constexpr size_t O_WHHT1 = O_WIH1  + SZ_WH;
constexpr size_t O_WHHS1 = O_WHHT1 + SZ_WH;
// h_t state, DOUBLE-BUFFERED by time parity:
// bf16 A-tiles [par 2][layer 2][bt 4][j 25][kt 16][lane 32][16]
constexpr size_t O_HT    = O_WHHS1 + SZ_WH;
constexpr size_t SZ_HT   = (size_t)2 * 2 * 4 * J_ * KT_H * TILE_B;
// c_t state: f32 C-tiles [layer 2][bt 4][j 25][ht 32][lane 32][8]
constexpr size_t O_CT    = O_HT + SZ_HT;
constexpr size_t SZ_CT   = (size_t)2 * 4 * J_ * 32 * TILE_B;
// x pre-swizzled to bf16 A-tiles: [t 64][j 25][bt 4][kt 4][lane 32][16]
constexpr size_t O_X     = O_CT + SZ_CT;
constexpr size_t SZ_X    = (size_t)T_ * J_ * 4 * KT_D * TILE_B;
} // namespace

// ---------------------------------------------------------------------------
// Pre-swizzle a fp32 [K, 2560] weight matrix into bf16 WMMA B-operand tiles.
// Tile (nt,kt): lane L holds column N = 16nt + L%16, K values
// k = 32kt + (q&7) + 8*(L>=16) + 16*(q>=8), q = 0..15   (ISA 16-bit B-matrix layout).
// ---------------------------------------------------------------------------
__global__ __launch_bounds__(256)
void stlstm_swizzle_w(const float* __restrict__ W, __bf16* __restrict__ outw, int KT)
{
    int tid = blockIdx.x * blockDim.x + threadIdx.x;
    int total = NT_ * KT * 32;
    if (tid >= total) return;
    int lane = tid & 31;
    int tile = tid >> 5;            // nt*KT + kt
    int kt = tile % KT;
    int nt = tile / KT;
    int N = (nt << 4) + (lane & 15);
    int kbase = (kt << 5) + ((lane >> 4) << 3);
    v16bf v;
#pragma unroll
    for (int q = 0; q < 16; ++q)
        v[q] = (__bf16)W[(size_t)(kbase + (q & 7) + ((q >> 3) << 4)) * NG_ + N];
    ((v16bf*)outw)[tid] = v;
}

// ---------------------------------------------------------------------------
// Pre-swizzle x [T,J,B,D] fp32 into bf16 WMMA A-operand tiles per (t,j,bt).
// A-tile layout: lane L holds row m = L%16; element q holds
// k = 32kt + (q&7) + 8*(L>=16) + 16*(q>=8)   (ISA 16-bit A-matrix layout).
// ---------------------------------------------------------------------------
__global__ __launch_bounds__(256)
void stlstm_swizzle_x(const float* __restrict__ x, __bf16* __restrict__ outx)
{
    int tid = blockIdx.x * blockDim.x + threadIdx.x;
    int total = T_ * J_ * 4 * KT_D * 32;
    if (tid >= total) return;
    int lane = tid & 31;
    int tile = tid >> 5;            // ((tj*4 + bt)*KT_D + kt)
    int kt = tile & (KT_D - 1);
    int rest = tile >> 2;
    int bt = rest & 3;
    int tj = rest >> 2;
    int m = lane & 15, half = lane >> 4;
    const float* xp = x + ((size_t)tj * B_ + 16 * bt + m) * D_ + (kt << 5);
    v16bf v;
#pragma unroll
    for (int q = 0; q < 16; ++q) {
        const int k = (q & 7) + (half << 3) + ((q & 8) ? 16 : 0);
        v[q] = (__bf16)xp[k];
    }
    ((v16bf*)outx)[tid] = v;
}

__device__ __forceinline__ float sigmoidf_(float x)
{
    return 1.0f / (1.0f + __expf(-x));
}

// ---------------------------------------------------------------------------
// Software-pipelined GEMM slice: acc[b] += A x W(nt = 32b + ht) over KT
// K-tiles.  Double-buffers the A fragment and the 5 B fragments one K-step
// ahead so global/LDS load latency overlaps the WMMA issue.
// ---------------------------------------------------------------------------
template <int KT>
__device__ __forceinline__ void mm5(const v16bf* __restrict__ A,
                                    const v16bf* __restrict__ Wb,
                                    v8f (&acc)[5], int ht, int lane)
{
    // tile (nt,kt) at v16bf index ((b*32 + ht)*KT + kt)*32 + lane
    const v16bf* bp = Wb + ((size_t)ht * KT) * 32 + lane;
    constexpr size_t BSTRIDE = (size_t)32 * KT * 32;   // per gate-block b

    v16bf a_cur = A[lane];
    v16bf b_cur[5];
#pragma unroll
    for (int b = 0; b < 5; ++b)
        b_cur[b] = bp[(size_t)b * BSTRIDE];

#pragma unroll 2
    for (int kt = 0; kt < KT - 1; ++kt) {
        v16bf a_nxt = A[(kt + 1) * 32 + lane];
        v16bf b_nxt[5];
#pragma unroll
        for (int b = 0; b < 5; ++b)
            b_nxt[b] = bp[(size_t)b * BSTRIDE + (kt + 1) * 32];
#pragma unroll
        for (int b = 0; b < 5; ++b)
            acc[b] = __builtin_amdgcn_wmma_f32_16x16x32_bf16(
                false, a_cur, false, b_cur[b], (short)0, acc[b], false, false);
        a_cur = a_nxt;
#pragma unroll
        for (int b = 0; b < 5; ++b) b_cur[b] = b_nxt[b];
    }
    // epilogue: last K-tile
#pragma unroll
    for (int b = 0; b < 5; ++b)
        acc[b] = __builtin_amdgcn_wmma_f32_16x16x32_bf16(
            false, a_cur, false, b_cur[b], (short)0, acc[b], false, false);
}

// ---------------------------------------------------------------------------
// LSTM cell elementwise update for one 16x16 h-tile (gate accumulators in
// C-layout: lane L, elem e -> m = e + 8*(L/16), n = 16*ht + L%16).
// Adds bias here, writes new h in A-operand swizzle to LDS h_s[nxt] and
// global h_t[write-parity], c to registers + global c_t.
// ---------------------------------------------------------------------------
template <bool WRITE_OUT>
__device__ __forceinline__ void cell5(
    v8f acc[5], float csr[8], const float bias[5],
    float*  __restrict__ ctg,    // c_t tiles base for (l,bt,j)
    __bf16* __restrict__ hs,     // LDS h_s[nxt] A-tiles
    __bf16* __restrict__ htg,    // global h_t A-tiles (write parity) for (l,bt,j)
    float*  __restrict__ outp,   // &out[((t*J+j)*B + 16*bt) * H]  (layer 1)
    int ht, int lane)
{
    const int lo = lane & 15, hi = lane >> 4;
    float* ctp = ctg + ((size_t)ht * 32 + lane) * 8;
    const int ktt    = ht >> 1;
    const int jb     = (lo & 7) + ((ht & 1) << 3);
    const int lane2b = (hi << 3) + ((lo >> 3) << 4);
#pragma unroll
    for (int e = 0; e < 8; ++e) {
        float ft = acc[0][e] + bias[0];
        float fs = acc[1][e] + bias[1];
        float ig = acc[2][e] + bias[2];
        float og = acc[3][e] + bias[3];
        float gg = acc[4][e] + bias[4];
        float cp = ctp[e];
        float c  = sigmoidf_(ig) * tanhf(gg)
                 + sigmoidf_(ft) * cp
                 + sigmoidf_(fs) * csr[e];
        float h  = sigmoidf_(og) * tanhf(c);
        csr[e] = c;
        ctp[e] = c;
        const int idx = ktt * 512 + (e + lane2b) * 16 + jb;  // A-swizzle elem idx
        __bf16 hb = (__bf16)h;
        hs[idx]  = hb;
        htg[idx] = hb;
        if (WRITE_OUT) {
            outp[(size_t)(e + (hi << 3)) * H_ + ((ht << 4) + lo)] = h;
        }
    }
}

// ---------------------------------------------------------------------------
// Fused 2-layer ST-LSTM recurrence.  Grid = 4 workgroups (one 16-row batch
// tile each), 512 threads = 16 waves.  Wave w processes h-tiles w and 16+w in
// two sequential passes (5 live v8f accumulators each).  h_s ping-pongs in
// LDS by step parity; h_t ping-pongs in global by time parity; c_s stays in
// registers (wave-private).  One barrier per (t,j) step.
// ---------------------------------------------------------------------------
__global__ __launch_bounds__(512, 1)
void stlstm_main(const float* __restrict__ b0,
                 const float* __restrict__ b1,
                 unsigned char* __restrict__ ws,
                 float* __restrict__ out)
{
    __shared__ __align__(32) __bf16 h0s[2][KT_H * 512];  // 2 x 16 KB
    __shared__ __align__(32) __bf16 h1s[2][KT_H * 512];  // 2 x 16 KB

    const int tid  = threadIdx.x;
    const int w    = tid >> 5;
    const int lane = tid & 31;
    const int lo   = lane & 15;
    const int bt   = blockIdx.x;

    const v16bf* Wih0  = (const v16bf*)(ws + O_WIH0);
    const v16bf* Whht0 = (const v16bf*)(ws + O_WHHT0);
    const v16bf* Whhs0 = (const v16bf*)(ws + O_WHHS0);
    const v16bf* Wih1  = (const v16bf*)(ws + O_WIH1);
    const v16bf* Whht1 = (const v16bf*)(ws + O_WHHT1);
    const v16bf* Whhs1 = (const v16bf*)(ws + O_WHHS1);
    __bf16* htst = (__bf16*)(ws + O_HT);
    float*  ctst = (float*) (ws + O_CT);
    const v16bf* xws = (const v16bf*)(ws + O_X);

    // h_t A-tile element base for (par, l, this bt, j)
    auto htbase = [&](int par, int l, int j) -> __bf16* {
        return htst + (size_t)((((par * 2 + l) * 4 + bt) * J_) + j) * (KT_H * 512);
    };

    // ---- zero initial state ----
    for (int i = tid; i < KT_H * 512; i += 512) {
        h0s[0][i] = (__bf16)0.0f; h0s[1][i] = (__bf16)0.0f;
        h1s[0][i] = (__bf16)0.0f; h1s[1][i] = (__bf16)0.0f;
    }
    {
        v4f z = {0.0f, 0.0f, 0.0f, 0.0f};
#pragma unroll
        for (int l = 0; l < 2; ++l) {
            v4f* ph = (v4f*)htbase(0, l, 0);          // zero parity-0 buffer only
            const int nh = J_ * KT_H * 512 / 8;
            for (int i = tid; i < nh; i += 512) ph[i] = z;
            v4f* pc = (v4f*)(ctst + (size_t)(l * 4 + bt) * J_ * 32 * 256);
            const int nc = J_ * 32 * 256 / 4;
            for (int i = tid; i < nc; i += 512) pc[i] = z;
        }
    }
    __syncthreads();

    // ---- per-wave bias preload: pass p owns ht = 16p + w ----
    float bias0v[2][5], bias1v[2][5];
#pragma unroll
    for (int p = 0; p < 2; ++p)
#pragma unroll
        for (int b = 0; b < 5; ++b) {
            const int n = ((b * 32 + (p << 4) + w) << 4) + lo;
            bias0v[p][b] = b0[n];
            bias1v[p][b] = b1[n];
        }

    float cs0[2][8], cs1[2][8];
#pragma unroll
    for (int p = 0; p < 2; ++p)
#pragma unroll
        for (int e = 0; e < 8; ++e) { cs0[p][e] = 0.0f; cs1[p][e] = 0.0f; }

#pragma unroll 1
    for (int s = 0; s < T_ * J_; ++s) {
        const int t   = s / J_;
        const int j   = s - t * J_;
        const int cur = s & 1, nxt = cur ^ 1;
        const int tp  = t & 1, tpn = tp ^ 1;

        const v16bf* xA = xws + ((size_t)(t * J_ + j) * 4 + bt) * (KT_D * 32);

        // ================= layer 0 (two passes) =================
        const v16bf* h0tr = (const v16bf*)htbase(tp, 0, j);
        __bf16*      h0tw = htbase(tpn, 0, j);
        float*       c0t  = ctst + (size_t)(0 * 4 + bt) * J_ * 32 * 256
                                 + (size_t)j * 32 * 256;
#pragma unroll 1
        for (int pass = 0; pass < 2; ++pass) {
            const int ht = (pass << 4) + w;
            v8f acc[5] = {};
            mm5<KT_D>(xA,                     Wih0,  acc, ht, lane);  // x   @ W_ih0
            mm5<KT_H>(h0tr,                   Whht0, acc, ht, lane);  // h_t @ W_hht0
            mm5<KT_H>((const v16bf*)h0s[cur], Whhs0, acc, ht, lane);  // h_s @ W_hhs0
            cell5<false>(acc, cs0[pass], bias0v[pass],
                         c0t, h0s[nxt], h0tw, nullptr, ht, lane);
        }
        __syncthreads();   // h0s[nxt] (layer-0 output) complete & visible

        // ================= layer 1 (two passes) =================
        const v16bf* h1tr = (const v16bf*)htbase(tp, 1, j);
        __bf16*      h1tw = htbase(tpn, 1, j);
        float*       c1t  = ctst + (size_t)(1 * 4 + bt) * J_ * 32 * 256
                                 + (size_t)j * 32 * 256;
        float*       outp = out + (size_t)((t * J_ + j) * B_ + 16 * bt) * H_;
#pragma unroll 1
        for (int pass = 0; pass < 2; ++pass) {
            const int ht = (pass << 4) + w;
            v8f acc[5] = {};
            mm5<KT_H>((const v16bf*)h0s[nxt], Wih1,  acc, ht, lane);  // h0  @ W_ih1
            mm5<KT_H>(h1tr,                   Whht1, acc, ht, lane);  // h_t @ W_hht1
            mm5<KT_H>((const v16bf*)h1s[cur], Whhs1, acc, ht, lane);  // h_s @ W_hhs1
            cell5<true>(acc, cs1[pass], bias1v[pass],
                        c1t, h1s[nxt], h1tw, outp, ht, lane);
        }
        // no end barrier needed: buffers ping-pong; the next step's mid
        // barrier fences h1s[nxt]/h1t writes before their first readers
    }
}

extern "C" void kernel_launch(void* const* d_in, const int* in_sizes, int n_in,
                              void* d_out, int out_size, void* d_ws, size_t ws_size,
                              hipStream_t stream)
{
    const float* x     = (const float*)d_in[0];
    const float* Wih0  = (const float*)d_in[1];
    const float* bih0  = (const float*)d_in[2];
    const float* Whht0 = (const float*)d_in[3];
    const float* Whhs0 = (const float*)d_in[4];
    const float* Wih1  = (const float*)d_in[5];
    const float* bih1  = (const float*)d_in[6];
    const float* Whht1 = (const float*)d_in[7];
    const float* Whhs1 = (const float*)d_in[8];
    unsigned char* ws  = (unsigned char*)d_ws;
    float* out         = (float*)d_out;

    auto sw = [&](const float* Wp, size_t off, int KT) {
        const int total = NT_ * KT * 32;
        stlstm_swizzle_w<<<(total + 255) / 256, 256, 0, stream>>>(
            Wp, (__bf16*)(ws + off), KT);
    };
    sw(Wih0,  O_WIH0,  KT_D);
    sw(Whht0, O_WHHT0, KT_H);
    sw(Whhs0, O_WHHS0, KT_H);
    sw(Wih1,  O_WIH1,  KT_H);
    sw(Whht1, O_WHHT1, KT_H);
    sw(Whhs1, O_WHHS1, KT_H);

    {
        const int total = T_ * J_ * 4 * KT_D * 32;
        stlstm_swizzle_x<<<(total + 255) / 256, 256, 0, stream>>>(
            x, (__bf16*)(ws + O_X));
    }

    stlstm_main<<<4, 512, 0, stream>>>(bih0, bih1, ws, out);
}
